// Encoder_71657234366478
// MI455X (gfx1250) — compile-verified, matrix operand order checked by reading, not run on hardware
//
#include <hip/hip_runtime.h>
#include <math.h>

typedef _Float16 half_t;
typedef __attribute__((ext_vector_type(16))) _Float16 v16h;
typedef __attribute__((ext_vector_type(8)))  _Float16 v8h;
typedef __attribute__((ext_vector_type(8)))  float    v8f;

#define N_NODES 16384
#define DEG     16
#define E_EDGES (N_NODES * DEG)
#define EPS_BN  1e-5f

// ---------------------------------------------------------------------------
// Fragment helpers
//
// 16-bit A-matrix 16x32 layout (CDNA5 ISA 7.12.2): lane l holds row m=l%16,
// lane-group g=l/16; half h (= vgpr v=h/2, half p=h%2) holds
//   k = kc*32 + (v<4 ? 2v+p : 16+2(v-4)+p) + 8g
// B (KxN) mirrors A with N on lanes. C/D: vgpr j -> row j+8g, lane -> col.
// ---------------------------------------------------------------------------

__device__ __host__ __forceinline__ int frag_k(int kc, int g, int hh) {
    int v = hh >> 1, p = hh & 1;
    int base = (v < 4) ? (2 * v + p) : (16 + 2 * (v - 4) + p);
    return kc * 32 + base + 8 * g;
}

// Load one B fragment (one 16x16 output tile, one 32-deep K chunk) from the
// pre-converted fragment buffer: tile stride = 512 halves, lane stride = 16.
__device__ __forceinline__ v16h load_frag_g(const half_t* __restrict__ tile, int lane) {
    return *(const v16h*)(tile + lane * 16);
}

// Assemble an A fragment from a row-major f16 activation row (LDS or global).
// rowbase points at element [m][0]; contiguous halves map straight onto the
// fragment striping: halves 0..7 <- row[kc*32+8g .. +7], 8..15 <- +16.
__device__ __forceinline__ v16h load_a_row(const half_t* __restrict__ rowbase, int kc, int g) {
    const half_t* p = rowbase + kc * 32 + g * 8;
    v8h lo = *(const v8h*)(p);
    v8h hi = *(const v8h*)(p + 16);
    v16h r;
#pragma unroll
    for (int j = 0; j < 8; ++j) { r[j] = lo[j]; r[8 + j] = hi[j]; }
    return r;
}

// ---------------------------------------------------------------------------
// Weight -> f16 fragment conversion
// ---------------------------------------------------------------------------
__global__ void k_frag(const float* __restrict__ W, half_t* __restrict__ out,
                       int K, int Ncols, int KC, int NT) {
    int idx = blockIdx.x * 256 + threadIdx.x;
    int total = KC * NT * 32;
    if (idx >= total) return;
    int lane = idx & 31, tile = idx >> 5;
    int kc = tile / NT, nt = tile % NT;
    int n = nt * 16 + (lane & 15);
    int g = lane >> 4;
    half_t* o = out + (size_t)tile * 512 + lane * 16;
#pragma unroll
    for (int hh = 0; hh < 16; ++hh) {
        int k = frag_k(kc, g, hh);
        float v = (k < K && n < Ncols) ? W[(size_t)k * Ncols + n] : 0.f;
        o[hh] = (half_t)v;
    }
}

// Concatenated EdgeConv weight [tw | tw+pw] -> fragments (Ncols = 2*OUT)
__global__ void k_frag_cat(const float* __restrict__ tw, const float* __restrict__ pw,
                           half_t* __restrict__ out, int K, int OUT, int KC) {
    int NT = (2 * OUT) / 16;
    int idx = blockIdx.x * 256 + threadIdx.x;
    int total = KC * NT * 32;
    if (idx >= total) return;
    int lane = idx & 31, tile = idx >> 5;
    int kc = tile / NT, nt = tile % NT;
    int n = nt * 16 + (lane & 15);
    int g = lane >> 4;
    half_t* o = out + (size_t)tile * 512 + lane * 16;
#pragma unroll
    for (int hh = 0; hh < 16; ++hh) {
        int k = frag_k(kc, g, hh);
        float v = 0.f;
        if (k < K) {
            if (n < OUT) v = tw[(size_t)k * OUT + n];
            else         v = tw[(size_t)k * OUT + (n - OUT)] + pw[(size_t)k * OUT + (n - OUT)];
        }
        o[hh] = (half_t)v;
    }
}

// ---------------------------------------------------------------------------
// Fused edge-MLP + NNConv (+ mean aggregate).  One wave32 per node: the
// node's 16 in-edges are the WMMA M dimension.  Never materializes We.
// ---------------------------------------------------------------------------
template <int KC, int NT, int WIN, int WOUT>
__device__ __forceinline__ void mlp_layer(half_t* __restrict__ A,
                                          const half_t* __restrict__ frag,
                                          const float* __restrict__ bias, int lane) {
    const int m = lane & 15, g = lane >> 4;
    v16h areg[KC];
#pragma unroll
    for (int kc = 0; kc < KC; ++kc) areg[kc] = load_a_row(A + m * WIN, kc, g);
    __syncthreads();
#pragma unroll
    for (int nt = 0; nt < NT; ++nt) {
        v8f c = {};
#pragma unroll
        for (int kc = 0; kc < KC; ++kc)
            c = __builtin_amdgcn_wmma_f32_16x16x32_f16(
                    false, areg[kc], false,
                    load_frag_g(frag + (size_t)(kc * NT + nt) * 512, lane),
                    (short)0, c, false, false);
        float bv = bias[nt * 16 + m];
        int col = nt * 16 + m;
#pragma unroll
        for (int j = 0; j < 8; ++j) {
            float v = c[j] + bv;
            v = fmaxf(v, 0.f);                       // relu
            A[(j + 8 * g) * WOUT + col] = (half_t)v; // row-major f16 act
        }
    }
    __syncthreads();
}

__global__ __launch_bounds__(128) void k_fused_nnconv(
    const float* __restrict__ h, const float* __restrict__ efeat,
    const int* __restrict__ src,
    const half_t* __restrict__ f0, const half_t* __restrict__ f1,
    const half_t* __restrict__ f2, const half_t* __restrict__ f3,
    const half_t* __restrict__ f4,
    const float* __restrict__ b0, const float* __restrict__ b1,
    const float* __restrict__ b2, const float* __restrict__ b3,
    const float* __restrict__ b4, const float* __restrict__ nnb,
    half_t* __restrict__ x16) {
    __shared__ half_t act[4][16 * 256];  // per-wave private activation tile
    __shared__ float  hb[4][16 * 8];     // h[src[e], 0..6] per edge row

    const int wave = threadIdx.x >> 5;
    const int lane = threadIdx.x & 31;
    const int node = blockIdx.x * 4 + wave;
    const int m = lane & 15, g = lane >> 4;
    const int e0 = node * DEG;

    // gather source-node features for the NNConv contraction
    if (g == 0) {
        int s = src[e0 + m];
#pragma unroll
        for (int i = 0; i < 8; ++i)
            hb[wave][m * 8 + i] = (i < 7) ? h[(size_t)s * 7 + i] : 0.f;
    }

    // A0: 16 edges x 3 features, zero-padded to K=32
    v16h a0;
#pragma unroll
    for (int i = 0; i < 16; ++i) a0[i] = (half_t)0.f;
    if (g == 0) {
        const float* ef = efeat + (size_t)(e0 + m) * 3;
        a0[0] = (half_t)ef[0]; a0[1] = (half_t)ef[1]; a0[2] = (half_t)ef[2];
    }

    half_t* A = act[wave];

    // layer0: 3 -> 256 (16 N-tiles, 1 K-chunk)
#pragma unroll
    for (int nt = 0; nt < 16; ++nt) {
        v8f c = {};
        c = __builtin_amdgcn_wmma_f32_16x16x32_f16(
                false, a0, false, load_frag_g(f0 + (size_t)nt * 512, lane),
                (short)0, c, false, false);
        float bv = b0[nt * 16 + m];
        int col = nt * 16 + m;
#pragma unroll
        for (int j = 0; j < 8; ++j) {
            float v = fmaxf(c[j] + bv, 0.f);
            A[(j + 8 * g) * 256 + col] = (half_t)v;
        }
    }
    __syncthreads();

    mlp_layer<8, 8, 256, 128>(A, f1, b1, lane);  // 256 -> 128
    mlp_layer<4, 4, 128, 64>(A, f2, b2, lane);   // 128 -> 64
    mlp_layer<2, 2, 64, 32>(A, f3, b3, lane);    // 64  -> 32

    // layer4 (32 -> 896, sigmoid) fused with NNConv:
    // column block i (128 wide) == We[:, i, :]
    v16h a4 = load_a_row(A + m * 32, 0, g);
    v8f msg[8] = {};
#pragma unroll
    for (int i = 0; i < 7; ++i) {
        float hv[8];
#pragma unroll
        for (int j = 0; j < 8; ++j) hv[j] = hb[wave][(j + 8 * g) * 8 + i];
#pragma unroll
        for (int t = 0; t < 8; ++t) {
            v8f c = {};
            c = __builtin_amdgcn_wmma_f32_16x16x32_f16(
                    false, a4, false,
                    load_frag_g(f4 + (size_t)(i * 8 + t) * 512, lane),
                    (short)0, c, false, false);
            float bv = b4[i * 128 + t * 16 + m];
#pragma unroll
            for (int j = 0; j < 8; ++j) {
                float z = c[j] + bv;
                float s = 1.f / (1.f + __expf(-z));  // sigmoid
                msg[t][j] += hv[j] * s;
            }
        }
    }

    // mean over the node's 16 edges (8 rows per lane-group + cross-group sum)
#pragma unroll
    for (int t = 0; t < 8; ++t) {
        float s = 0.f;
#pragma unroll
        for (int j = 0; j < 8; ++j) s += msg[t][j];
        s += __shfl_xor(s, 16, 32);
        float xv = s * (1.f / 16.f) + nnb[t * 16 + m];
        if (g == 0) x16[(size_t)node * 128 + t * 16 + m] = (half_t)xv;
    }
}

// ---------------------------------------------------------------------------
// EdgeConv pass A: per-node GEMM  x[N,KIN] @ [tw | tw+pw]  ->  u, w (f32)
// ---------------------------------------------------------------------------
template <int KIN, int OUT>
__global__ __launch_bounds__(128) void k_econv_gemm(
    const half_t* __restrict__ x16, const half_t* __restrict__ frag,
    const float* __restrict__ tb, const float* __restrict__ pb,
    float* __restrict__ u, float* __restrict__ w) {
    constexpr int KC = KIN / 32;
    constexpr int NT = (2 * OUT) / 16;
    const int wave = threadIdx.x >> 5, lane = threadIdx.x & 31;
    const int tile = blockIdx.x * 4 + wave;
    const int m = lane & 15, g = lane >> 4;

    const half_t* row = x16 + (size_t)(tile * 16 + m) * KIN;
    v16h areg[KC];
#pragma unroll
    for (int kc = 0; kc < KC; ++kc) areg[kc] = load_a_row(row, kc, g);

#pragma unroll
    for (int nt = 0; nt < NT; ++nt) {
        v8f c = {};
#pragma unroll
        for (int kc = 0; kc < KC; ++kc)
            c = __builtin_amdgcn_wmma_f32_16x16x32_f16(
                    false, areg[kc], false,
                    load_frag_g(frag + (size_t)(kc * NT + nt) * 512, lane),
                    (short)0, c, false, false);
        int col = nt * 16 + m;
        if (nt < OUT / 16) {  // u = x@tw  (no bias)
#pragma unroll
            for (int j = 0; j < 8; ++j)
                u[(size_t)(tile * 16 + j + 8 * g) * OUT + col] = c[j];
        } else {              // w = x@(tw+pw) + (tb+pb)
            int cc = col - OUT;
            float bv = tb[cc] + pb[cc];
#pragma unroll
            for (int j = 0; j < 8; ++j)
                w[(size_t)(tile * 16 + j + 8 * g) * OUT + cc] = c[j] + bv;
        }
    }
}

// ---------------------------------------------------------------------------
// EdgeConv pass B: deterministic BN stats (per-block partials, then reduce)
// ---------------------------------------------------------------------------
template <int OUT>
__global__ __launch_bounds__(256) void k_stats(
    const float* __restrict__ u, const float* __restrict__ w,
    const int* __restrict__ src, float* __restrict__ part) {
    constexpr int ROWS = 256 / OUT;
    constexpr int EPB = E_EDGES / 256;  // edges per block
    __shared__ float s1[256], s2[256];
    const int c = threadIdx.x % OUT;
    const int r = threadIdx.x / OUT;
    const long ebase = (long)blockIdx.x * EPB;
    float sum = 0.f, sq = 0.f;
    for (int e = r; e < EPB; e += ROWS) {
        long ee = ebase + e;
        int node = (int)(ee >> 4);
        float mv = w[(size_t)node * OUT + c] - u[(size_t)src[ee] * OUT + c];
        sum += mv; sq += mv * mv;
    }
    s1[threadIdx.x] = sum; s2[threadIdx.x] = sq;
    __syncthreads();
    if (r == 0) {
#pragma unroll
        for (int rr = 1; rr < ROWS; ++rr) { sum += s1[rr * OUT + c]; sq += s2[rr * OUT + c]; }
        part[(size_t)blockIdx.x * (2 * OUT) + c] = sum;
        part[(size_t)blockIdx.x * (2 * OUT) + OUT + c] = sq;
    }
}

__global__ void k_reduce_stats(const float* __restrict__ part, float* __restrict__ stats, int n2) {
    int c = threadIdx.x;
    if (c >= n2) return;
    float s = 0.f;
    for (int b = 0; b < 256; ++b) s += part[(size_t)b * n2 + c];
    stats[c] = s;
}

// ---------------------------------------------------------------------------
// EdgeConv pass C: normalize + segment_max (gathers hit L2-resident u table)
// ---------------------------------------------------------------------------
template <int OUT, bool F16OUT>
__global__ __launch_bounds__(256) void k_norm_max(
    const float* __restrict__ u, const float* __restrict__ w,
    const int* __restrict__ src, const float* __restrict__ stats,
    const float* __restrict__ gam, const float* __restrict__ bet,
    float* __restrict__ out32, half_t* __restrict__ out16) {
    long idx = (long)blockIdx.x * 256 + threadIdx.x;
    int node = (int)(idx / OUT);
    int c = (int)(idx % OUT);
    float mu = stats[c] * (1.f / E_EDGES);
    float var = stats[OUT + c] * (1.f / E_EDGES) - mu * mu;
    float scale = gam[c] * rsqrtf(var + EPS_BN);
    float shift = bet[c] - mu * scale;
    float wv = w[(size_t)node * OUT + c];
    const int* sp = src + node * DEG;
    float mx = -INFINITY;
#pragma unroll
    for (int k = 0; k < DEG; ++k) {
        float mval = wv - u[(size_t)sp[k] * OUT + c];
        mx = fmaxf(mx, mval * scale + shift);
    }
    if constexpr (F16OUT) out16[idx] = (half_t)mx;
    else                  out32[idx] = mx;
}

// ---------------------------------------------------------------------------
// Launch
// ---------------------------------------------------------------------------
extern "C" void kernel_launch(void* const* d_in, const int* in_sizes, int n_in,
                              void* d_out, int out_size, void* d_ws, size_t ws_size,
                              hipStream_t stream) {
    (void)in_sizes; (void)n_in; (void)out_size; (void)ws_size;
    const float* h     = (const float*)d_in[0];
    const float* efeat = (const float*)d_in[1];
    const int*   src   = (const int*)d_in[2];
    // d_in[3] = dst (implicit: dst[e] == e/16)
    const float* mw[5] = {(const float*)d_in[4], (const float*)d_in[6], (const float*)d_in[8],
                          (const float*)d_in[10], (const float*)d_in[12]};
    const float* mb[5] = {(const float*)d_in[5], (const float*)d_in[7], (const float*)d_in[9],
                          (const float*)d_in[11], (const float*)d_in[13]};
    const float* nnb = (const float*)d_in[14];
    const float* tw[3] = {(const float*)d_in[15], (const float*)d_in[21], (const float*)d_in[27]};
    const float* tb[3] = {(const float*)d_in[16], (const float*)d_in[22], (const float*)d_in[28]};
    const float* pw[3] = {(const float*)d_in[17], (const float*)d_in[23], (const float*)d_in[29]};
    const float* pb[3] = {(const float*)d_in[18], (const float*)d_in[24], (const float*)d_in[30]};
    const float* gg[3] = {(const float*)d_in[19], (const float*)d_in[25], (const float*)d_in[31]};
    const float* bb[3] = {(const float*)d_in[20], (const float*)d_in[26], (const float*)d_in[32]};

    char* ws = (char*)d_ws;
    size_t off = 0;
    auto take = [&](size_t bytes) -> void* {
        void* p = ws + off;
        off += (bytes + 255) & ~(size_t)255;
        return p;
    };

    half_t* f0  = (half_t*)take(16 * 1024);    // 1x16 tiles
    half_t* f1  = (half_t*)take(64 * 1024);    // 8x8
    half_t* f2  = (half_t*)take(16 * 1024);    // 4x4
    half_t* f3  = (half_t*)take(4 * 1024);     // 2x2
    half_t* f4  = (half_t*)take(56 * 1024);    // 1x56
    half_t* fe1 = (half_t*)take(32 * 1024);    // 4x8
    half_t* fe2 = (half_t*)take(8 * 1024);     // 2x4
    half_t* fe3 = (half_t*)take(2 * 1024);     // 1x2
    half_t* x0  = (half_t*)take((size_t)N_NODES * 128 * 2);
    half_t* x1  = (half_t*)take((size_t)N_NODES * 64 * 2);
    half_t* x2  = (half_t*)take((size_t)N_NODES * 32 * 2);
    float*  ub  = (float*)take((size_t)N_NODES * 64 * 4);
    float*  wb  = (float*)take((size_t)N_NODES * 64 * 4);
    float*  part = (float*)take(256 * 128 * 4);
    float*  stats = (float*)take(128 * 4);

    // weight -> fragment conversion (tiny; runs every call, deterministic)
    k_frag<<<2, 256, 0, stream>>>(mw[0], f0, 3, 256, 1, 16);
    k_frag<<<8, 256, 0, stream>>>(mw[1], f1, 256, 128, 8, 8);
    k_frag<<<2, 256, 0, stream>>>(mw[2], f2, 128, 64, 4, 4);
    k_frag<<<1, 256, 0, stream>>>(mw[3], f3, 64, 32, 2, 2);
    k_frag<<<7, 256, 0, stream>>>(mw[4], f4, 32, 896, 1, 56);
    k_frag_cat<<<4, 256, 0, stream>>>(tw[0], pw[0], fe1, 128, 64, 4);
    k_frag_cat<<<1, 256, 0, stream>>>(tw[1], pw[1], fe2, 64, 32, 2);
    k_frag_cat<<<1, 256, 0, stream>>>(tw[2], pw[2], fe3, 32, 16, 1);

    // fused edge-MLP + NNConv (dominant: ~42 GFLOP of f16 WMMA)
    k_fused_nnconv<<<N_NODES / 4, 128, 0, stream>>>(
        h, efeat, src, f0, f1, f2, f3, f4,
        mb[0], mb[1], mb[2], mb[3], mb[4], nnb, x0);

    // EdgeConv layer 1: 128 -> 64
    k_econv_gemm<128, 64><<<N_NODES / 64, 128, 0, stream>>>(x0, fe1, tb[0], pb[0], ub, wb);
    k_stats<64><<<256, 256, 0, stream>>>(ub, wb, src, part);
    k_reduce_stats<<<1, 128, 0, stream>>>(part, stats, 128);
    k_norm_max<64, true><<<(N_NODES * 64) / 256, 256, 0, stream>>>(
        ub, wb, src, stats, gg[0], bb[0], nullptr, x1);

    // EdgeConv layer 2: 64 -> 32
    k_econv_gemm<64, 32><<<N_NODES / 64, 128, 0, stream>>>(x1, fe2, tb[1], pb[1], ub, wb);
    k_stats<32><<<256, 256, 0, stream>>>(ub, wb, src, part);
    k_reduce_stats<<<1, 64, 0, stream>>>(part, stats, 64);
    k_norm_max<32, true><<<(N_NODES * 32) / 256, 256, 0, stream>>>(
        ub, wb, src, stats, gg[1], bb[1], nullptr, x2);

    // EdgeConv layer 3: 32 -> 16 (final, f32 to d_out)
    k_econv_gemm<32, 16><<<N_NODES / 64, 128, 0, stream>>>(x2, fe3, tb[2], pb[2], ub, wb);
    k_stats<16><<<256, 256, 0, stream>>>(ub, wb, src, part);
    k_reduce_stats<<<1, 32, 0, stream>>>(part, stats, 32);
    k_norm_max<16, false><<<(N_NODES * 16) / 256, 256, 0, stream>>>(
        ub, wb, src, stats, gg[2], bb[2], (float*)d_out, nullptr);
}
